// AttentionHead_19825569039039
// MI455X (gfx1250) — matrix operand order repaired and checked by reading, not compiled, not run on hardware
//
#include <hip/hip_runtime.h>
#include <hip/hip_bf16.h>

typedef __attribute__((ext_vector_type(16))) __bf16 v16bf;
typedef __attribute__((ext_vector_type(8)))  __bf16 v8bf;
typedef __attribute__((ext_vector_type(8)))  float  v8f;
typedef __attribute__((ext_vector_type(8)))  float  v8ff;

#define SEQ   4096
#define DDIM  1024
#define DK    128
#define KSTR  136     // K LDS row stride (128 + 8 pad), 272B = 17*16B
#define VSTR  72      // V LDS row stride (64 + 8 pad), 144B = 9*16B
#define PSTR  72      // P staging row stride

#define GLOBAL_AS __attribute__((address_space(1)))

#define WMMA_BF16(A,B,C) \
  __builtin_amdgcn_wmma_f32_16x16x32_bf16(false, (A), false, (B), (short)0, (C), false, false)

// ---- explicit global-address-space accessors ------------------------------
static __device__ __forceinline__ v8bf gload8(const __bf16* p) {
  return *(const GLOBAL_AS v8bf*)(unsigned long long)p;
}
static __device__ __forceinline__ v8ff gload8f(const float* p) {
  return *(const GLOBAL_AS v8ff*)(unsigned long long)p;
}
static __device__ __forceinline__ float gloadf(const float* p) {
  return *(const GLOBAL_AS float*)(unsigned long long)p;
}
static __device__ __forceinline__ void gstore_h(__bf16* p, __bf16 v) {
  *(GLOBAL_AS __bf16*)(unsigned long long)p = v;
}
static __device__ __forceinline__ void gstore8h(__bf16* p, v8bf v) {
  *(GLOBAL_AS v8bf*)(unsigned long long)p = v;
}
static __device__ __forceinline__ void gstore_f(float* p, float v) {
  *(GLOBAL_AS float*)(unsigned long long)p = v;
}

static __device__ __forceinline__ v16bf frag_cat(v8bf lo, v8bf hi) {
  return __builtin_shufflevector(lo, hi, 0,1,2,3,4,5,6,7,8,9,10,11,12,13,14,15);
}
// A-operand (16-bit, 16x32): lane L row (L&15); e -> K = kbase + (e>>3)*16 + h*8 + (e&7)
static __device__ __forceinline__ v16bf a_frag(const __bf16* row, int kbase, int h) {
  return frag_cat(gload8(row + kbase + h * 8), gload8(row + kbase + 16 + h * 8));
}
// B-operand (16-bit, 32x16): lane L col (L&15); e -> K = kbase + (e>>3)*16 + h*8 + (e&7)
static __device__ __forceinline__ v16bf b_frag(const __bf16* row, int kbase, int h) {
  return frag_cat(gload8(row + kbase + h * 8), gload8(row + kbase + 16 + h * 8));
}
// Same fragment shapes out of LDS
static __device__ __forceinline__ v16bf lds_b_frag(const __bf16* row, int kbase, int h) {
  return frag_cat(*reinterpret_cast<const v8bf*>(row + kbase + h * 8),
                  *reinterpret_cast<const v8bf*>(row + kbase + 16 + h * 8));
}

// ---------------------------------------------------------------------------
// Kernel 0a: Wq/Wk/Wv [1024,128] f32 -> W^T [128,1024] bf16
// ---------------------------------------------------------------------------
__global__ __launch_bounds__(256) void wtrans(const float* __restrict__ Wq,
                                              const float* __restrict__ Wk,
                                              const float* __restrict__ Wv,
                                              __bf16* __restrict__ WqT,
                                              __bf16* __restrict__ WkT,
                                              __bf16* __restrict__ WvT) {
  int i = blockIdx.x * blockDim.x + threadIdx.x;
  if (i >= DK * DDIM) return;
  int kk = i >> 10, d = i & 1023;
  int src = d * DK + kk;
  gstore_h(WqT + i, (__bf16)gloadf(Wq + src));
  gstore_h(WkT + i, (__bf16)gloadf(Wk + src));
  gstore_h(WvT + i, (__bf16)gloadf(Wv + src));
}

// ---------------------------------------------------------------------------
// Kernel 0b: x f32 -> bf16, one pass
// ---------------------------------------------------------------------------
__global__ __launch_bounds__(256) void xcvt(const float* __restrict__ x,
                                            __bf16* __restrict__ xb) {
  size_t i = ((size_t)blockIdx.x * blockDim.x + threadIdx.x) * 8;
  v8ff f = gload8f(x + i);
  v8bf o;
#pragma unroll
  for (int j = 0; j < 8; ++j) o[j] = (__bf16)f[j];
  gstore8h(xb + i, o);
}

// ---------------------------------------------------------------------------
// Kernel 1: q,k,v projections. One wave per 32-row (2-tile) slab; each W
// B-fragment feeds two WMMAs.
// ---------------------------------------------------------------------------
static __device__ __forceinline__ void proj_tile2(const __bf16* xrow0, const __bf16* xrow1,
                                                  const __bf16* WT, int l16, int h,
                                                  v8f accA[8], v8f accB[8]) {
#pragma unroll
  for (int nt = 0; nt < 8; ++nt)
#pragma unroll
    for (int r = 0; r < 8; ++r) { accA[nt][r] = 0.0f; accB[nt][r] = 0.0f; }
  for (int c = 0; c < 32; ++c) {
    v16bf a0 = a_frag(xrow0, c * 32, h);
    v16bf a1 = a_frag(xrow1, c * 32, h);
#pragma unroll
    for (int nt = 0; nt < 8; ++nt) {
      v16bf bf = b_frag(WT + (size_t)(nt * 16 + l16) * DDIM, c * 32, h);
      accA[nt] = WMMA_BF16(a0, bf, accA[nt]);
      accB[nt] = WMMA_BF16(a1, bf, accB[nt]);
    }
  }
}

__global__ __launch_bounds__(128) void qkv_proj(const __bf16* __restrict__ xb,
                                                const __bf16* __restrict__ WqT,
                                                const __bf16* __restrict__ WkT,
                                                const __bf16* __restrict__ WvT,
                                                __bf16* __restrict__ qb,
                                                __bf16* __restrict__ kb,
                                                __bf16* __restrict__ vT) {
  const int lane = threadIdx.x & 31;
  const int w    = threadIdx.x >> 5;
  const int l16  = lane & 15;
  const int h    = lane >> 4;
  const int tw   = blockIdx.x * 4 + w;                 // 0..511
  const size_t base = (size_t)tw * 32;                 // first of 32 rows
  const __bf16* xrow0 = xb + (base + (size_t)l16) * DDIM;
  const __bf16* xrow1 = xrow0 + (size_t)16 * DDIM;

  constexpr float qscale = 1.4426950408889634f * 0.08838834764831845f; // log2e/sqrt(dk)
  v8f accA[8], accB[8];

  proj_tile2(xrow0, xrow1, WqT, l16, h, accA, accB);
#pragma unroll
  for (int nt = 0; nt < 8; ++nt)
#pragma unroll
    for (int r = 0; r < 8; ++r) {
      gstore_h(qb + (base + r + 8 * h) * DK + nt * 16 + l16, (__bf16)(accA[nt][r] * qscale));
      gstore_h(qb + (base + 16 + r + 8 * h) * DK + nt * 16 + l16, (__bf16)(accB[nt][r] * qscale));
    }

  proj_tile2(xrow0, xrow1, WkT, l16, h, accA, accB);
#pragma unroll
  for (int nt = 0; nt < 8; ++nt)
#pragma unroll
    for (int r = 0; r < 8; ++r) {
      gstore_h(kb + (base + r + 8 * h) * DK + nt * 16 + l16, (__bf16)accA[nt][r]);
      gstore_h(kb + (base + 16 + r + 8 * h) * DK + nt * 16 + l16, (__bf16)accB[nt][r]);
    }

  proj_tile2(xrow0, xrow1, WvT, l16, h, accA, accB);
  {
    const int bb  = (int)(base >> 12);
    const int sin = (int)(base & 4095);
#pragma unroll
    for (int nt = 0; nt < 8; ++nt)
#pragma unroll
      for (int r = 0; r < 8; ++r) {
        __bf16* col = vT + ((size_t)bb * DK + nt * 16 + l16) * SEQ;
        gstore_h(col + sin + r + 8 * h,      (__bf16)accA[nt][r]);
        gstore_h(col + sin + 16 + r + 8 * h, (__bf16)accB[nt][r]);
      }
  }
}

// ---------------------------------------------------------------------------
// Kernel 2: flash attention. Workgroup (4 waves) = 64 query rows of one batch;
// 64-key K/V block cooperatively staged in LDS; block i+1 in staging
// registers during compute(i); block i+2 prefetched into L2.
// ---------------------------------------------------------------------------
__global__ __launch_bounds__(128) void attn_flash(const __bf16* __restrict__ q,
                                                  const __bf16* __restrict__ k,
                                                  const __bf16* __restrict__ vT,
                                                  float* __restrict__ out) {
  __shared__ __align__(16) __bf16 Kbuf[64 * KSTR];     // 17408 B
  __shared__ __align__(16) __bf16 Vbuf[DK * VSTR];     // 18432 B
  __shared__ __align__(16) __bf16 Pbuf[4][16 * PSTR];  //  9216 B

  const int tid  = threadIdx.x;
  const int lane = tid & 31;
  const int w    = tid >> 5;
  const int l16  = lane & 15;
  const int h    = lane >> 4;
  const int b    = blockIdx.x >> 6;                    // 64 blocks per batch
  const int q0   = (blockIdx.x & 63) * 64;
  const size_t qrow0 = (size_t)b * SEQ + q0 + w * 16;

  const __bf16* kbase = k  + (size_t)b * SEQ * DK;
  const __bf16* vbase = vT + (size_t)b * DK * SEQ;

  const int krow  = tid >> 1;
  const int kpart = (tid & 1) * 64;

  const __bf16* qrow = q + (qrow0 + (size_t)l16) * DK;
  v16bf qa[4];
#pragma unroll
  for (int c = 0; c < 4; ++c) qa[c] = a_frag(qrow, c * 32, h);

  v8f acc[8];
#pragma unroll
  for (int nt = 0; nt < 8; ++nt)
#pragma unroll
    for (int r = 0; r < 8; ++r) acc[nt][r] = 0.0f;
  float rmax[8], rsum[8];
#pragma unroll
  for (int r = 0; r < 8; ++r) { rmax[r] = -3.0e38f; rsum[r] = 0.0f; }

  v8bf kst[8], vst[8];                                 // register staging
  auto load_block = [&](int kblk) {
#pragma unroll
    for (int j = 0; j < 8; ++j)
      kst[j] = gload8(kbase + (size_t)(kblk + krow) * DK + kpart + j * 8);
#pragma unroll
    for (int j = 0; j < 8; ++j)
      vst[j] = gload8(vbase + (size_t)tid * SEQ + kblk + j * 8);
  };
  auto store_block = [&]() {
    __bf16* Kd = Kbuf + krow * KSTR + kpart;
#pragma unroll
    for (int j = 0; j < 8; ++j) *reinterpret_cast<v8bf*>(Kd + j * 8) = kst[j];
    __bf16* Vd = Vbuf + tid * VSTR;
#pragma unroll
    for (int j = 0; j < 8; ++j) *reinterpret_cast<v8bf*>(Vd + j * 8) = vst[j];
  };

  load_block(0);
  store_block();
  __syncthreads();

  __bf16* lp = &Pbuf[w][0];
  constexpr int NBLK = SEQ / 64;

  for (int i = 0; i < NBLK; ++i) {
    const bool more = (i + 1) < NBLK;
    if (more) load_block((i + 1) * 64);                // latency hides behind WMMAs
    if (i + 2 < NBLK) {                                // pull block i+2 toward L2
      __builtin_prefetch(kbase + (size_t)((i + 2) * 64 + krow) * DK + kpart, 0, 0);
      __builtin_prefetch(vbase + (size_t)tid * SEQ + (i + 2) * 64, 0, 0);
    }

    // scores (log2 domain): S[16q x 64m] = Q . K^T, K fragments from LDS
    v8f s[4];
#pragma unroll
    for (int t = 0; t < 4; ++t)
#pragma unroll
      for (int r = 0; r < 8; ++r) s[t][r] = 0.0f;
#pragma unroll
    for (int c = 0; c < 4; ++c) {
      v16bf aq = qa[c];
#pragma unroll
      for (int t = 0; t < 4; ++t)
        s[t] = WMMA_BF16(aq, lds_b_frag(Kbuf + (t * 16 + l16) * KSTR, c * 32, h), s[t]);
    }

    // online softmax (base-2); collect correction factors per row, then apply
    v8f corrv;
    __bf16* lrow = lp + 8 * h * PSTR + l16;
#pragma unroll
    for (int r = 0; r < 8; ++r) {
      float mx = fmaxf(fmaxf(s[0][r], s[1][r]), fmaxf(s[2][r], s[3][r]));
#pragma unroll
      for (int m = 8; m >= 1; m >>= 1) mx = fmaxf(mx, __shfl_xor(mx, m, 32));
      const float mnew = fmaxf(rmax[r], mx);
      const float corr = exp2f(rmax[r] - mnew);
      corrv[r] = corr;
      rmax[r] = mnew;
      const float p0 = exp2f(s[0][r] - mnew);
      const float p1 = exp2f(s[1][r] - mnew);
      const float p2 = exp2f(s[2][r] - mnew);
      const float p3 = exp2f(s[3][r] - mnew);
      float ps = (p0 + p1) + (p2 + p3);
#pragma unroll
      for (int m = 8; m >= 1; m >>= 1) ps += __shfl_xor(ps, m, 32);
      rsum[r] = rsum[r] * corr + ps;
      __bf16* lr = lrow + r * PSTR;                    // C-layout row r+8h
      lr[0]  = (__bf16)p0;
      lr[16] = (__bf16)p1;
      lr[32] = (__bf16)p2;
      lr[48] = (__bf16)p3;
    }
#pragma unroll
    for (int nt = 0; nt < 8; ++nt) acc[nt] *= corrv;   // vectorized rescale

    // P back in A-operand layout (same-wave DS ordering covers the RAW)
    const __bf16* pr = lp + l16 * PSTR;
    v16bf pa0 = frag_cat(*reinterpret_cast<const v8bf*>(pr + h * 8),
                         *reinterpret_cast<const v8bf*>(pr + 16 + h * 8));
    v16bf pa1 = frag_cat(*reinterpret_cast<const v8bf*>(pr + 32 + h * 8),
                         *reinterpret_cast<const v8bf*>(pr + 48 + h * 8));

    // Z += P . V, V fragments from LDS
#pragma unroll
    for (int nt = 0; nt < 8; ++nt) {
      const __bf16* vrow = Vbuf + (nt * 16 + l16) * VSTR;
      acc[nt] = WMMA_BF16(pa0, lds_b_frag(vrow, 0,  h), acc[nt]);
      acc[nt] = WMMA_BF16(pa1, lds_b_frag(vrow, 32, h), acc[nt]);
    }

    __syncthreads();
    if (more) {
      store_block();
      __syncthreads();
    }
  }

  // epilogue: normalize, write f32 [B,S,128]
#pragma unroll
  for (int r = 0; r < 8; ++r) {
    const float inv = 1.0f / rsum[r];
    const size_t row = qrow0 + r + 8 * h;
#pragma unroll
    for (int nt = 0; nt < 8; ++nt)
      gstore_f(out + row * DK + nt * 16 + l16, acc[nt][r] * inv);
  }
}

// ---------------------------------------------------------------------------
extern "C" void kernel_launch(void* const* d_in, const int* in_sizes, int n_in,
                              void* d_out, int out_size, void* d_ws, size_t ws_size,
                              hipStream_t stream) {
  (void)in_sizes; (void)n_in; (void)out_size; (void)ws_size;
  const float* x  = (const float*)d_in[0];
  const float* Wq = (const float*)d_in[1];
  const float* Wk = (const float*)d_in[2];
  const float* Wv = (const float*)d_in[3];
  float* out = (float*)d_out;

  __bf16* WqT = (__bf16*)d_ws;
  __bf16* WkT = WqT + (size_t)DK * DDIM;
  __bf16* WvT = WkT + (size_t)DK * DDIM;
  __bf16* qb  = WvT + (size_t)DK * DDIM;
  __bf16* kb  = qb + (size_t)4 * SEQ * DK;
  __bf16* vT  = kb + (size_t)4 * SEQ * DK;
  __bf16* xb  = vT + (size_t)4 * SEQ * DK;

  hipLaunchKernelGGL(wtrans, dim3(512), dim3(256), 0, stream, Wq, Wk, Wv, WqT, WkT, WvT);
  hipLaunchKernelGGL(xcvt, dim3((unsigned)((size_t)4 * SEQ * DDIM / 8 / 256)), dim3(256),
                     0, stream, x, xb);
  hipLaunchKernelGGL(qkv_proj, dim3(128), dim3(128), 0, stream,
                     xb, WqT, WkT, WvT, qb, kb, vT);
  hipLaunchKernelGGL(attn_flash, dim3(256), dim3(128), 0, stream, qb, kb, vT, out);
}
